// SSMaxMultiheadAttention_72636486910098
// MI455X (gfx1250) — matrix operand order repaired and checked by reading, not compiled, not run on hardware
//
#include <hip/hip_runtime.h>
#include <hip/hip_bf16.h>

#define BB 4
#define SS 2048
#define EE 512
#define HH 8
#define DD 64

typedef __attribute__((ext_vector_type(16))) __bf16 v16bf;
typedef __attribute__((ext_vector_type(8)))  __bf16 v8bf;
typedef __attribute__((ext_vector_type(8)))  float  v8f;
typedef __attribute__((ext_vector_type(4)))  float  v4f;

__device__ __forceinline__ v8f zero8() {
  v8f z = {0.f, 0.f, 0.f, 0.f, 0.f, 0.f, 0.f, 0.f};
  return z;
}

__device__ __forceinline__ v16bf cat8(v8bf lo, v8bf hi) {
  return __builtin_shufflevector(lo, hi, 0, 1, 2, 3, 4, 5, 6, 7, 8, 9, 10, 11,
                                 12, 13, 14, 15);
}

__device__ __forceinline__ v8f wmma_bf(v16bf a, v16bf b, v8f c) {
  // D = A(16x32 bf16) * B(32x16 bf16) + C(16x16 f32)
  return __builtin_amdgcn_wmma_f32_16x16x32_bf16(false, a, false, b, (short)0, c,
                                                 false, false);
}

// A fragment 16x32 bf16 from row-major source (ld in elements).
// ISA layout: lane<16 holds row=lane&15, K={kh*8..+7, 16+kh*8..+7}, kh=lane>>4.
__device__ __forceinline__ v16bf load_a_bf16(const __bf16* p, int ld) {
  const int lane = threadIdx.x & 31;
  const int r = lane & 15;
  const int kh = (lane >> 4) << 3;
  const __bf16* row = p + (size_t)r * ld + kh;
  v8bf lo = *(const v8bf*)row;          // 16B load
  v8bf hi = *(const v8bf*)(row + 16);   // 16B load
  return cat8(lo, hi);
}

// B fragment 32x16 where B[k][n] = p[n*ld + k] (source rows are B columns).
// lane holds col n=lane&15, K={kh*8..+7, 16+kh*8..+7}, kh=lane>>4.
__device__ __forceinline__ v16bf load_bT_bf16(const __bf16* p, int ld) {
  const int lane = threadIdx.x & 31;
  const int n = lane & 15;
  const int kb = (lane >> 4) << 3;
  const __bf16* q = p + (size_t)n * ld + kb;
  return cat8(*(const v8bf*)q, *(const v8bf*)(q + 16));
}

// ---------------------------------------------------------------------------
// Kernel 0a: elementwise fp32 -> bf16 (activations). 8 elements/thread.
// ---------------------------------------------------------------------------
__global__ void cvt_f32_bf16_kernel(const float* __restrict__ x,
                                    __bf16* __restrict__ y, int n8) {
  const int i = blockIdx.x * blockDim.x + threadIdx.x;
  if (i < n8) {
    v8f v = *(const v8f*)(x + (size_t)i * 8);
    *(v8bf*)(y + (size_t)i * 8) = __builtin_convertvector(v, v8bf);
  }
}

// ---------------------------------------------------------------------------
// Kernel 0b: weight convert + transpose: WT[n][k] = bf16(W[k][n]).
// blockIdx.y selects among the 4 weight matrices.
// ---------------------------------------------------------------------------
__global__ void wT_kernel(const float* __restrict__ W0,
                          const float* __restrict__ W1,
                          const float* __restrict__ W2,
                          const float* __restrict__ W3,
                          __bf16* __restrict__ T0, __bf16* __restrict__ T1,
                          __bf16* __restrict__ T2, __bf16* __restrict__ T3) {
  const int w = blockIdx.y;
  const float* W = (w == 0) ? W0 : (w == 1) ? W1 : (w == 2) ? W2 : W3;
  __bf16* T = (w == 0) ? T0 : (w == 1) ? T1 : (w == 2) ? T2 : T3;
  const int t = blockIdx.x * blockDim.x + threadIdx.x;  // 0..32767
  const int n = t >> 6;
  const int k8 = (t & 63) * 8;
  __bf16* o = T + (size_t)n * EE + k8;
#pragma unroll
  for (int e = 0; e < 8; ++e) o[e] = (__bf16)W[(size_t)(k8 + e) * EE + n];
}

// ---------------------------------------------------------------------------
// Kernel 1: fused QKV projection. Xbf[8192x512] @ WT + b -> bf16.
// Q,K stored [B,H,S,D]; V stored transposed [B,H,D,S].
// Block = 256 threads (8 waves); wave owns one 16-row tile x 128 cols.
// ---------------------------------------------------------------------------
__global__ void qkv_proj_kernel(const __bf16* __restrict__ Xq,
                                const __bf16* __restrict__ Xk,
                                const __bf16* __restrict__ Xv,
                                const __bf16* __restrict__ WqT,
                                const __bf16* __restrict__ WkT,
                                const __bf16* __restrict__ WvT,
                                const float* __restrict__ bq,
                                const float* __restrict__ bk,
                                const float* __restrict__ bv,
                                __bf16* __restrict__ Qbf,
                                __bf16* __restrict__ Kbf,
                                __bf16* __restrict__ Vt) {
  const int which = blockIdx.z;
  const __bf16* X   = (which == 0) ? Xq : (which == 1) ? Xk : Xv;
  const __bf16* WT  = (which == 0) ? WqT : (which == 1) ? WkT : WvT;
  const float* bias = (which == 0) ? bq : (which == 1) ? bk : bv;
  __bf16* dst       = (which == 0) ? Qbf : (which == 1) ? Kbf : Vt;

  const int wave = threadIdx.x >> 5;
  const int mbase = (blockIdx.x * 8 + wave) * 16;
  const int nbase = blockIdx.y * 128;

  v8f acc[8];
#pragma unroll
  for (int t = 0; t < 8; ++t) acc[t] = zero8();

  for (int k0 = 0; k0 < EE; k0 += 32) {
    v16bf a = load_a_bf16(X + (size_t)mbase * EE + k0, EE);
#pragma unroll
    for (int t = 0; t < 8; ++t) {
      // B[k][n] = WT[nbase+t*16+n][k0+k] : contiguous per lane
      v16bf bm = load_bT_bf16(WT + (size_t)(nbase + t * 16) * EE + k0, EE);
      acc[t] = wmma_bf(a, bm, acc[t]);
    }
  }

  const int lane = threadIdx.x & 31;
  const int crow = (lane >> 4) << 3;   // C/D layout: M = f + 8*(lane>=16)
  const int ccol = lane & 15;          // N = lane % 16
#pragma unroll
  for (int t = 0; t < 8; ++t) {
    const int n = nbase + t * 16 + ccol;
    const float bv_ = bias[n];
    const int h = n >> 6, d = n & 63;
#pragma unroll
    for (int f = 0; f < 8; ++f) {
      const int m = mbase + f + crow;
      const int b = m >> 11, s = m & (SS - 1);
      const float val = acc[t][f] + bv_;
      if (which < 2)
        dst[((size_t)(b * HH + h) * SS + s) * DD + d] = (__bf16)val;
      else
        dst[((size_t)(b * HH + h) * DD + d) * SS + s] = (__bf16)val;
    }
  }
}

// ---------------------------------------------------------------------------
// Kernel 2: attention with SSMax. One workgroup per (b,h, 32-query rows).
// Phase 1: WMMA scores -> 256 KB LDS panel (CDNA5 320KB WGP LDS); each K
//          fragment is loaded once and feeds both q-subtile WMMAs.
// Phase 2: wave-per-row softmax(x*0.125 + s*log S) with wave32 shfl_xor
//          reductions; fp32 weights stream to d_out exactly once as fully
//          coalesced 512B-per-wave float4 stores (the ~537 MB that sets the
//          kernel's bandwidth floor).
// Phase 3: WMMA P@V from LDS -> bf16 attn_out [B,S,E].
// ---------------------------------------------------------------------------
__global__ void ssmax_attn_kernel(const __bf16* __restrict__ Qbf,
                                  const __bf16* __restrict__ Kbf,
                                  const __bf16* __restrict__ Vt,
                                  const float* __restrict__ s_scalar,
                                  float* __restrict__ attnW,
                                  __bf16* __restrict__ AO) {
  extern __shared__ float sc[];   // 32 x 2048 fp32 score panel = 256 KB

  const int bh = blockIdx.y;            // b*H + h
  const int qbase = blockIdx.x * 32;
  const int tid = threadIdx.x;
  const int wave = tid >> 5;
  const int lane = tid & 31;

  const __bf16* Qp = Qbf + ((size_t)bh * SS + qbase) * DD;
  const __bf16* Kp = Kbf + (size_t)bh * SS * DD;
  const __bf16* Vp = Vt + (size_t)bh * DD * SS;

  const float shift = s_scalar[0] * logf((float)SS);
  const float scale = 0.125f;           // 1/sqrt(64)

  // Preload the 2 q-subtile A fragments (2 K-steps each, D=64).
  v16bf qa[2][2];
#pragma unroll
  for (int qt = 0; qt < 2; ++qt)
#pragma unroll
    for (int kk = 0; kk < 2; ++kk)
      qa[qt][kk] = load_a_bf16(Qp + (qt * 16) * DD + kk * 32, DD);

  const int crow = (lane >> 4) << 3;
  const int ccol = lane & 15;

  // ---- Phase 1: scores (scaled + SSMax shift) into LDS ----
  // One wave per 16-key tile; each K fragment feeds both q-subtiles.
  for (int jt = wave; jt < 128; jt += 8) {
    if (jt + 8 < 128)
      __builtin_prefetch((const void*)(Kp + (size_t)((jt + 8) * 16) * DD), 0, 0);
    v8f acc0 = zero8(), acc1 = zero8();
#pragma unroll
    for (int kk = 0; kk < 2; ++kk) {
      v16bf bm = load_bT_bf16(Kp + (size_t)(jt * 16) * DD + kk * 32, DD);
      acc0 = wmma_bf(qa[0][kk], bm, acc0);
      acc1 = wmma_bf(qa[1][kk], bm, acc1);
    }
#pragma unroll
    for (int f = 0; f < 8; ++f) {
      sc[(f + crow) * SS + jt * 16 + ccol]        = acc0[f] * scale + shift;
      sc[(16 + f + crow) * SS + jt * 16 + ccol]   = acc1[f] * scale + shift;
    }
  }
  __syncthreads();

  // ---- Phase 2: softmax, one wave per row (4 rows/wave), coalesced IO ----
  for (int r = wave; r < 32; r += 8) {
    v4f* rp4 = (v4f*)(sc + (size_t)r * SS);   // 512 float4s per row

    float lmax = -3.0e38f;
#pragma unroll 4
    for (int i = 0; i < 16; ++i) {
      v4f v = rp4[lane + i * 32];
      lmax = fmaxf(lmax, fmaxf(fmaxf(v[0], v[1]), fmaxf(v[2], v[3])));
    }
#pragma unroll
    for (int off = 16; off > 0; off >>= 1)
      lmax = fmaxf(lmax, __shfl_xor(lmax, off, 32));

    float lsum = 0.f;
#pragma unroll 4
    for (int i = 0; i < 16; ++i) {
      v4f v = rp4[lane + i * 32];
      v4f t;
#pragma unroll
      for (int c = 0; c < 4; ++c) t[c] = __expf(v[c] - lmax);
      rp4[lane + i * 32] = t;                  // keep exp in LDS
      lsum += (t[0] + t[1]) + (t[2] + t[3]);
    }
#pragma unroll
    for (int off = 16; off > 0; off >>= 1)
      lsum += __shfl_xor(lsum, off, 32);

    const float inv = 1.0f / lsum;
    v4f* op4 = (v4f*)(attnW + (size_t)bh * SS * SS + (size_t)(qbase + r) * SS);
#pragma unroll 4
    for (int i = 0; i < 16; ++i) {
      v4f w = rp4[lane + i * 32] * inv;
      rp4[lane + i * 32] = w;                  // normalized P stays for P@V
      op4[lane + i * 32] = w;                  // 512B contiguous per wave
    }
  }
  __syncthreads();

  // ---- Phase 3: P @ V. 8 waves = 2 q-subtiles x 4 d-subtiles ----
  {
    const int qt = wave >> 2;
    const int dt = wave & 3;
    v8f acc = zero8();
    const int r = lane & 15;
    const int kh = (lane >> 4) << 3;
    for (int jt = 0; jt < 64; ++jt) {    // K=32 per step over 2048 keys
      if (jt + 4 < 64)
        __builtin_prefetch(
            (const void*)(Vp + (size_t)(dt * 16) * SS + (jt + 4) * 32), 0, 0);
      const float* rowp = sc + (qt * 16 + r) * SS + jt * 32 + kh;
      v8f lo = *(const v8f*)rowp;        // ds 32B loads
      v8f hi = *(const v8f*)(rowp + 16);
      v16bf a = cat8(__builtin_convertvector(lo, v8bf),
                     __builtin_convertvector(hi, v8bf));
      v16bf bm = load_bT_bf16(Vp + (size_t)(dt * 16) * SS + jt * 32, SS);
      acc = wmma_bf(a, bm, acc);
    }
    const int b = bh >> 3, h = bh & 7;
#pragma unroll
    for (int f = 0; f < 8; ++f) {
      const int srow = qbase + qt * 16 + f + crow;
      AO[(size_t)(b * SS + srow) * EE + h * DD + dt * 16 + ccol] =
          (__bf16)acc[f];
    }
  }
}

// ---------------------------------------------------------------------------
// Kernel 3: output projection. AO[8192x512]bf16 @ WoT + bo -> fp32 out.
// ---------------------------------------------------------------------------
__global__ void out_proj_kernel(const __bf16* __restrict__ AO,
                                const __bf16* __restrict__ WoT,
                                const float* __restrict__ bo,
                                float* __restrict__ out) {
  const int wave = threadIdx.x >> 5;
  const int mbase = (blockIdx.x * 8 + wave) * 16;
  const int nbase = blockIdx.y * 128;

  v8f acc[8];
#pragma unroll
  for (int t = 0; t < 8; ++t) acc[t] = zero8();

  for (int k0 = 0; k0 < EE; k0 += 32) {
    v16bf a = load_a_bf16(AO + (size_t)mbase * EE + k0, EE);
#pragma unroll
    for (int t = 0; t < 8; ++t) {
      v16bf bm = load_bT_bf16(WoT + (size_t)(nbase + t * 16) * EE + k0, EE);
      acc[t] = wmma_bf(a, bm, acc[t]);
    }
  }

  const int lane = threadIdx.x & 31;
  const int crow = (lane >> 4) << 3;
  const int ccol = lane & 15;
#pragma unroll
  for (int t = 0; t < 8; ++t) {
    const int n = nbase + t * 16 + ccol;
    const float bv_ = bo[n];
#pragma unroll
    for (int f = 0; f < 8; ++f) {
      const int m = mbase + f + crow;
      out[(size_t)m * EE + n] = acc[t][f] + bv_;
    }
  }
}

extern "C" void kernel_launch(void* const* d_in, const int* in_sizes, int n_in,
                              void* d_out, int out_size, void* d_ws, size_t ws_size,
                              hipStream_t stream) {
  const float* query = (const float*)d_in[0];
  const float* key_  = (const float*)d_in[1];
  const float* value = (const float*)d_in[2];
  const float* Wq = (const float*)d_in[3];
  const float* bq = (const float*)d_in[4];
  const float* Wk = (const float*)d_in[5];
  const float* bk = (const float*)d_in[6];
  const float* Wv = (const float*)d_in[7];
  const float* bv = (const float*)d_in[8];
  const float* Wo = (const float*)d_in[9];
  const float* bo = (const float*)d_in[10];
  const float* sS = (const float*)d_in[11];

  const size_t NME = (size_t)BB * SS * EE;   // 8192*512 = 4,194,304
  const size_t NW  = (size_t)EE * EE;        // 262,144
  __bf16* p = (__bf16*)d_ws;
  __bf16* Xq = p; p += NME;
  __bf16* Xk = p; p += NME;
  __bf16* Xv = p; p += NME;
  __bf16* WqT = p; p += NW;
  __bf16* WkT = p; p += NW;
  __bf16* WvT = p; p += NW;
  __bf16* WoT = p; p += NW;
  __bf16* Qbf = p; p += NME;
  __bf16* Kbf = p; p += NME;
  __bf16* Vt  = p; p += NME;
  __bf16* AO  = p; p += NME;

  float* out   = (float*)d_out;              // [B,S,E] first
  float* attnW = out + NME;                  // then [B,H,S,S]

  // K0a: activations fp32 -> bf16 (8 elems/thread)
  const int n8 = (int)(NME / 8);
  cvt_f32_bf16_kernel<<<dim3((n8 + 255) / 256), dim3(256), 0, stream>>>(query, Xq, n8);
  cvt_f32_bf16_kernel<<<dim3((n8 + 255) / 256), dim3(256), 0, stream>>>(key_, Xk, n8);
  cvt_f32_bf16_kernel<<<dim3((n8 + 255) / 256), dim3(256), 0, stream>>>(value, Xv, n8);
  // K0b: weights fp32 -> bf16 transposed
  wT_kernel<<<dim3(128, 4), dim3(256), 0, stream>>>(Wq, Wk, Wv, Wo, WqT, WkT,
                                                    WvT, WoT);

  // K1: QKV projections (3 matrices x 512 M-tiles x 4 N-blocks)
  qkv_proj_kernel<<<dim3(64, 4, 3), dim3(256), 0, stream>>>(
      Xq, Xk, Xv, WqT, WkT, WvT, bq, bk, bv, Qbf, Kbf, Vt);

  // K2: attention; 256 KB dynamic LDS score panel per WG (320 KB WGP limit)
  const int lds_bytes = 32 * SS * (int)sizeof(float);
  (void)hipFuncSetAttribute((const void*)ssmax_attn_kernel,
                            hipFuncAttributeMaxDynamicSharedMemorySize, lds_bytes);
  ssmax_attn_kernel<<<dim3(SS / 32, BB * HH), dim3(256), lds_bytes, stream>>>(
      Qbf, Kbf, Vt, sS, attnW, AO);

  // K3: output projection
  out_proj_kernel<<<dim3(64, 4), dim3(256), 0, stream>>>(AO, WoT, bo, out);
}
